// PoseNet_60713657696678
// MI455X (gfx1250) — matrix operand order repaired and checked by reading, not compiled, not run on hardware
//
#include <hip/hip_runtime.h>
#include <hip/hip_bf16.h>
#include <stdint.h>

#define B_   4
#define DIM_ 3
#define D1_  4
#define N_   32768
#define L0_  16384
#define L1_  8192
#define L2_  4096
#define F_   15
#define CIN_ 68
#define SLOPE_ 0.1f

typedef __attribute__((ext_vector_type(16))) __bf16 v16bf;
typedef __attribute__((ext_vector_type(8)))  float  v8f;

union V16 { v16bf v; uint4 q[2]; };

__device__ __forceinline__ float bf2f(uint16_t h) {
    uint32_t u = ((uint32_t)h) << 16;
    return __builtin_bit_cast(float, u);
}
__device__ __forceinline__ uint16_t f2bf(float f) {
    uint32_t u = __builtin_bit_cast(uint32_t, f);
    uint32_t r = (u + 0x7fffu + ((u >> 16) & 1u)) >> 16;
    return (uint16_t)r;
}
__device__ __forceinline__ float lrelu(float x) { return x >= 0.f ? x : SLOPE_ * x; }

// ---------------------------------------------------------------------------
// CDNA5 async global->LDS copy (8-byte granules, ASYNCcnt-tracked).
// VDST operand = 32-bit LDS byte address, VADDR = 64-bit global address.
// ---------------------------------------------------------------------------
__device__ __forceinline__ void async_copy_b64(void* lds_ptr, const void* gptr) {
    uint32_t la = (uint32_t)(uintptr_t)lds_ptr;       // low 32 bits = LDS offset
    uint64_t ga = (uint64_t)(uintptr_t)gptr;
    asm volatile("global_load_async_to_lds_b64 %0, %1, off"
                 :: "v"(la), "v"(ga)
                 : "memory");
}
__device__ __forceinline__ void wait_async_all() {
    asm volatile("s_wait_asynccnt 0" ::: "memory");
}

// ---------------------------------------------------------------------------
// Pointwise conv block: 3 -> 32 -> 32 -> 64 with leaky relu, thread-per-point.
// ---------------------------------------------------------------------------
__global__ void feat_block_kernel(const float* __restrict__ pc,  // [B][3][N]
                                  const float* __restrict__ W1, const float* __restrict__ b1,
                                  const float* __restrict__ W2, const float* __restrict__ b2,
                                  const float* __restrict__ W3, const float* __restrict__ b3,
                                  float* __restrict__ out) {    // [B][64][N]
    __shared__ float sW1[32 * 3], sb1[32], sW2[32 * 32], sb2[32], sW3[64 * 32], sb3[64];
    for (int i = threadIdx.x; i < 96;   i += blockDim.x) sW1[i] = W1[i];
    for (int i = threadIdx.x; i < 32;   i += blockDim.x) sb1[i] = b1[i];
    for (int i = threadIdx.x; i < 1024; i += blockDim.x) sW2[i] = W2[i];
    for (int i = threadIdx.x; i < 32;   i += blockDim.x) sb2[i] = b2[i];
    for (int i = threadIdx.x; i < 2048; i += blockDim.x) sW3[i] = W3[i];
    for (int i = threadIdx.x; i < 64;   i += blockDim.x) sb3[i] = b3[i];
    __syncthreads();

    int idx = blockIdx.x * blockDim.x + threadIdx.x;   // exact grid = B*N
    int b = idx / N_, n = idx % N_;
    float x[3];
    for (int d = 0; d < 3; d++) x[d] = pc[((size_t)b * 3 + d) * N_ + n];
    float h1[32];
    for (int o = 0; o < 32; o++) {
        float s = sb1[o];
        for (int d = 0; d < 3; d++) s += sW1[o * 3 + d] * x[d];
        h1[o] = lrelu(s);
    }
    float h2[32];
    for (int o = 0; o < 32; o++) {
        float s = sb2[o];
        for (int c = 0; c < 32; c++) s += sW2[o * 32 + c] * h1[c];
        h2[o] = lrelu(s);
    }
    for (int o = 0; o < 64; o++) {
        float s = sb3[o];
        for (int c = 0; c < 32; c++) s += sW3[o * 32 + c] * h2[c];
        out[((size_t)b * 64 + o) * N_ + n] = lrelu(s);
    }
}

// ---------------------------------------------------------------------------
// Utility kernels
// ---------------------------------------------------------------------------
__global__ void zero_f32(float* __restrict__ p, long n) {
    long i = (long)blockIdx.x * blockDim.x + threadIdx.x;
    if (i < n) p[i] = 0.f;
}

__global__ void f32_to_bf16(const float* __restrict__ s, uint16_t* __restrict__ d, long n) {
    long i = (long)blockIdx.x * blockDim.x + threadIdx.x;
    if (i < n) d[i] = f2bf(s[i]);
}

// [B][M][L] fp32 -> [B][L][M] bf16 (site-major for contiguous gathers)
__global__ void to_site_major_bf16(const float* __restrict__ s, uint16_t* __restrict__ d,
                                   int M, int L) {
    long i = (long)blockIdx.x * blockDim.x + threadIdx.x;
    long tot = (long)B_ * M * L;
    if (i >= tot) return;
    int b = (int)(i / ((long)M * L));
    int r = (int)(i % ((long)M * L));
    int c = r / L, l = r % L;
    d[((size_t)b * L + l) * M + c] = f2bf(s[i]);
}

// Wb[MOUT][CIN][F] fp32 -> A[MOUT][KPAD] bf16, k = f*CIN + c, zero-padded.
__global__ void prep_A(const float* __restrict__ W, uint16_t* __restrict__ A,
                       int MOUT, int CIN, int KPAD) {
    int i = blockIdx.x * blockDim.x + threadIdx.x;
    if (i >= MOUT * KPAD) return;
    int o = i / KPAD, k = i % KPAD;
    float v = 0.f;
    if (k < F_ * CIN) {
        int f = k / CIN, c = k % CIN;
        v = W[((size_t)o * CIN + c) * F_ + f];
    }
    A[i] = f2bf(v);
}

// ---------------------------------------------------------------------------
// Splat: scatter (el ++ feat) * bary onto lattice sites with fp32 atomics.
// lat layout: [B][L][68] site-major.
// ---------------------------------------------------------------------------
__global__ void splat_kernel(const float* __restrict__ el,    // [B][4][M]
                             const float* __restrict__ feat,  // [B][64][M]
                             const float* __restrict__ bary,  // [B][4][M]
                             const int*   __restrict__ offs,  // [B][4][M]
                             float* __restrict__ lat,         // [B][L][68]
                             int M, int L) {
    int idx = blockIdx.x * blockDim.x + threadIdx.x;
    if (idx >= B_ * M) return;
    int b = idx / M, m = idx % M;
    float v[CIN_];
    for (int c = 0; c < 4;  c++) v[c]     = el[((size_t)b * 4 + c) * M + m];
    for (int c = 0; c < 64; c++) v[4 + c] = feat[((size_t)b * 64 + c) * M + m];
    for (int d = 0; d < 4; d++) {
        float w = bary[((size_t)b * 4 + d) * M + m];
        int   o = offs[((size_t)b * 4 + d) * M + m];
        float* dst = lat + ((size_t)b * L + o) * CIN_;
        for (int c = 0; c < CIN_; c++) atomicAdd(dst + c, v[c] * w);
    }
}

// ---------------------------------------------------------------------------
// Fused gather -> blur GEMM (WMMA bf16, f32 acc) -> bias/lrelu -> 1x1 conv
// (WMMA) -> bias/lrelu -> store.
// Block = 128 threads (4 waves); each wave owns 16 output columns.
// LDS: A[MOUT][KPAD] + W2[MOUT][MOUT] + Bpanel[64][KPAD] + H[64][MOUT] (bf16).
// Gathers go memory->LDS via GLOBAL_LOAD_ASYNC_TO_LDS_B64 (ASYNCcnt).
// ---------------------------------------------------------------------------
template <int CIN, int MOUT, int KPAD>
__global__ __launch_bounds__(128) void blur_kernel(
        const uint16_t* __restrict__ lat,    // [B][L][CIN] bf16 site-major
        const int*      __restrict__ nbr,    // [B][F][L]
        const uint16_t* __restrict__ A,      // [MOUT][KPAD] bf16
        const float*    __restrict__ bias1,  // [MOUT]
        const uint16_t* __restrict__ W2,     // [MOUT][MOUT] bf16
        const float*    __restrict__ bias2,  // [MOUT]
        float*          __restrict__ out,    // [B][MOUT][L]
        const uint16_t* __restrict__ scale,  // [B][L][CIN] or nullptr
        int L) {
    extern __shared__ char smem[];
    uint16_t* sA  = (uint16_t*)smem;               // MOUT*KPAD
    uint16_t* sW2 = sA + MOUT * KPAD;              // MOUT*MOUT
    uint16_t* sB  = sW2 + MOUT * MOUT;             // 64*KPAD
    uint16_t* sH  = sB + 64 * KPAD;                // 64*MOUT

    const int tid  = threadIdx.x;
    const int wave = tid >> 5;
    const int lane = tid & 31;
    const int tilesPerB = L >> 6;
    const int b    = blockIdx.x / tilesPerB;
    const int tile = blockIdx.x % tilesPerB;

    // Stage A and W2 into LDS via async DMA (full reuse across all 64 columns)
    for (int i = tid; i < MOUT * KPAD / 4; i += 128)
        async_copy_b64(sA + i * 4, A + (size_t)i * 4);
    for (int i = tid; i < MOUT * MOUT / 4; i += 128)
        async_copy_b64(sW2 + i * 4, W2 + (size_t)i * 4);

    // Gather this wave's B panel: 16 columns x F neighbor-site vectors.
    const int colBase = tile * 64 + wave * 16;
    uint16_t* myB = sB + wave * 16 * KPAD;
    const int qwords = CIN / 4;                    // CIN*2 bytes / 8
    for (int seg = lane; seg < 16 * F_; seg += 32) {
        int cl  = seg / F_;
        int f   = seg % F_;
        int col = colBase + cl;
        int idx = nbr[((size_t)b * F_ + f) * L + col];
        const uint16_t* s = lat + ((size_t)b * L + idx) * CIN;
        uint16_t* d = myB + cl * KPAD + f * CIN;
        if (scale) {  // correlation path: elementwise product with own column
            const uint32_t* sw = (const uint32_t*)s;
            const uint32_t* sc = (const uint32_t*)(scale + ((size_t)b * L + col) * CIN);
            uint32_t* dw = (uint32_t*)d;
            for (int i = 0; i < CIN / 2; i++) {
                uint32_t aw = sw[i], bw = sc[i];
                uint16_t lo = f2bf(bf2f((uint16_t)aw)         * bf2f((uint16_t)bw));
                uint16_t hi = f2bf(bf2f((uint16_t)(aw >> 16)) * bf2f((uint16_t)(bw >> 16)));
                dw[i] = ((uint32_t)hi << 16) | lo;
            }
        } else {      // pure gather: async memory->LDS, no VGPR round-trip
            for (int i = 0; i < qwords; i++)
                async_copy_b64(d + i * 4, s + i * 4);
        }
    }
    // Zero K-pad region per column
    for (int cl = lane; cl < 16; cl += 32) {
        uint32_t* d = (uint32_t*)(myB + cl * KPAD + F_ * CIN);
        for (int i = 0; i < (KPAD - F_ * CIN) / 2; i++) d[i] = 0u;
    }
    wait_async_all();
    __syncthreads();

    // ---------------- GEMM1: out[MOUT][16] = A(MOUT x K) * Bpanel(K x 16) ---
    constexpr int MT = MOUT / 16;
    v8f acc[MT];
#pragma unroll
    for (int mt = 0; mt < MT; mt++) acc[mt] = (v8f){0.f, 0.f, 0.f, 0.f, 0.f, 0.f, 0.f, 0.f};

    const int colL   = lane & 15;
    const int hiHalf = lane >> 4;        // 0/1
    const int aOff   = hiHalf * 16;      // A K sub-offset (contiguous, matches B)
    const int bOff   = hiHalf * 16;      // B K sub-offset (contiguous, matches A)
    const uint16_t* bCol = myB + colL * KPAD;

#pragma unroll 4
    for (int ks = 0; ks < KPAD / 32; ks++) {
        V16 bv;
        bv.q[0] = *(const uint4*)(bCol + ks * 32 + bOff);
        bv.q[1] = *(const uint4*)(bCol + ks * 32 + bOff + 8);
#pragma unroll
        for (int mt = 0; mt < MT; mt++) {
            const uint16_t* aRow = sA + (size_t)(mt * 16 + colL) * KPAD + ks * 32 + aOff;
            V16 av;
            av.q[0] = *(const uint4*)(aRow);
            av.q[1] = *(const uint4*)(aRow + 8);
            acc[mt] = __builtin_amdgcn_wmma_f32_16x16x32_bf16(
                false, av.v, false, bv.v, (short)0, acc[mt], false, false);
        }
    }

    // bias + lrelu; stage h (C/D layout -> site-major bf16 in LDS)
#pragma unroll
    for (int mt = 0; mt < MT; mt++) {
#pragma unroll
        for (int v = 0; v < 8; v++) {
            int Mrow = mt * 16 + hiHalf * 8 + v;
            float x = lrelu(acc[mt][v] + bias1[Mrow]);
            sH[(wave * 16 + colL) * MOUT + Mrow] = f2bf(x);
        }
    }
    __syncthreads();

    // ---------------- GEMM2: 1x1 conv, K = MOUT ----------------------------
    v8f acc2[MT];
#pragma unroll
    for (int mt = 0; mt < MT; mt++) acc2[mt] = (v8f){0.f, 0.f, 0.f, 0.f, 0.f, 0.f, 0.f, 0.f};
    const uint16_t* hCol = sH + (wave * 16 + colL) * MOUT;
#pragma unroll
    for (int ks = 0; ks < MOUT / 32; ks++) {
        V16 bv;
        bv.q[0] = *(const uint4*)(hCol + ks * 32 + bOff);
        bv.q[1] = *(const uint4*)(hCol + ks * 32 + bOff + 8);
#pragma unroll
        for (int mt = 0; mt < MT; mt++) {
            const uint16_t* aRow = sW2 + (size_t)(mt * 16 + colL) * MOUT + ks * 32 + aOff;
            V16 av;
            av.q[0] = *(const uint4*)(aRow);
            av.q[1] = *(const uint4*)(aRow + 8);
            acc2[mt] = __builtin_amdgcn_wmma_f32_16x16x32_bf16(
                false, av.v, false, bv.v, (short)0, acc2[mt], false, false);
        }
    }

    const int colG = colBase + colL;
#pragma unroll
    for (int mt = 0; mt < MT; mt++) {
#pragma unroll
        for (int v = 0; v < 8; v++) {
            int Mrow = mt * 16 + hiHalf * 8 + v;
            float x = lrelu(acc2[mt][v] + bias2[Mrow]);
            out[((size_t)b * MOUT + Mrow) * L + colG] = x;
        }
    }
}

// ---------------------------------------------------------------------------
// Mean pool over L2 for o1c | o2c | v  -> pooled (B,192)
// ---------------------------------------------------------------------------
__global__ void pool_kernel(const float* __restrict__ o1c, const float* __restrict__ o2c,
                            const float* __restrict__ v, float* __restrict__ pooled) {
    int b = blockIdx.x / 192, j = blockIdx.x % 192;
    const float* src; int ch;
    if (j < 64)       { src = o1c; ch = j; }
    else if (j < 128) { src = o2c; ch = j - 64; }
    else              { src = v;   ch = j - 128; }
    const float* p = src + ((size_t)b * 64 + ch) * L2_;
    float s = 0.f;
    for (int i = threadIdx.x; i < L2_; i += 256) s += p[i];
    __shared__ float red[256];
    red[threadIdx.x] = s;
    __syncthreads();
    for (int st = 128; st > 0; st >>= 1) {
        if (threadIdx.x < st) red[threadIdx.x] += red[threadIdx.x + st];
        __syncthreads();
    }
    if (threadIdx.x == 0) pooled[b * 192 + j] = red[0] * (1.f / L2_);
}

// ---------------------------------------------------------------------------
// FC head (B=4 -> tiny; plain fp32 dot products)
// ---------------------------------------------------------------------------
__global__ void fc_kernel(const float* __restrict__ in, const float* __restrict__ W,
                          const float* __restrict__ bias, float* __restrict__ out,
                          int K, int O) {
    int i = blockIdx.x * blockDim.x + threadIdx.x;
    if (i >= B_ * O) return;
    int b = i / O, o = i % O;
    const float* x = in + (size_t)b * K;
    const float* w = W + (size_t)o * K;
    float s = bias[o];
    for (int k = 0; k < K; k++) s += w[k] * x[k];
    out[i] = fmaxf(s, 0.f);
}

__global__ void fc3_kernel(const float* __restrict__ in, const float* __restrict__ W,
                           const float* __restrict__ bias, float* __restrict__ d_out) {
    int i = blockIdx.x * blockDim.x + threadIdx.x;
    if (i >= B_ * 6) return;
    int b = i / 6, j = i % 6;
    const float* x = in + (size_t)b * 2048;
    float s = bias[j];
    for (int k = 0; k < 2048; k++) s += W[(size_t)j * 2048 + k] * x[k];
    if (j < 3) d_out[b * 3 + j] = s;            // first return: pose[:, :3]
    else       d_out[12 + b * 3 + (j - 3)] = s; // second return: pose[:, 3:]
}

// ---------------------------------------------------------------------------
// Host orchestration
// ---------------------------------------------------------------------------
static inline size_t smem_bytes(int MOUT, int KPAD) {
    return (size_t)(MOUT * KPAD + MOUT * MOUT + 64 * KPAD + 64 * MOUT) * 2;
}

extern "C" void kernel_launch(void* const* d_in, const int* in_sizes, int n_in,
                              void* d_out, int out_size, void* d_ws, size_t ws_size,
                              hipStream_t stream) {
    (void)in_sizes; (void)n_in; (void)out_size; (void)ws_size;

    const float* pc[2] = {(const float*)d_in[0], (const float*)d_in[1]};
    const float* el[3][2]; const float* bary[3][2];
    const int* offs[3][2]; const int* nbr[3][2];
    int ii = 2;
    for (int s = 0; s < 3; s++)
        for (int p = 0; p < 2; p++) {
            el[s][p]   = (const float*)d_in[ii++];
            bary[s][p] = (const float*)d_in[ii++];
            offs[s][p] = (const int*)d_in[ii++];
            nbr[s][p]  = (const int*)d_in[ii++];
        }
    const int* corr_idx1 = (const int*)d_in[26];
    const int* corr_idx2 = (const int*)d_in[27];
    // 28,29: max_hash_cnt scalars == L2, unused
    const float* cbW1 = (const float*)d_in[30]; const float* cbb1 = (const float*)d_in[31];
    const float* cbW2 = (const float*)d_in[32]; const float* cbb2 = (const float*)d_in[33];
    const float* cbW3 = (const float*)d_in[34]; const float* cbb3 = (const float*)d_in[35];
    const float *bcnWb[3], *bcnbb[3], *bcnW1[3], *bcnb1[3];
    for (int i = 0; i < 3; i++) {
        bcnWb[i] = (const float*)d_in[36 + 4 * i];
        bcnbb[i] = (const float*)d_in[37 + 4 * i];
        bcnW1[i] = (const float*)d_in[38 + 4 * i];
        bcnb1[i] = (const float*)d_in[39 + 4 * i];
    }
    const float* Wc1 = (const float*)d_in[48]; const float* bc1 = (const float*)d_in[49];
    const float* Wc2 = (const float*)d_in[50]; const float* bc2 = (const float*)d_in[51];
    const float* Wv1 = (const float*)d_in[52]; const float* bv1 = (const float*)d_in[53];
    const float* Wv2 = (const float*)d_in[54]; const float* bv2 = (const float*)d_in[55];
    const float* fc1W = (const float*)d_in[56]; const float* fc1b = (const float*)d_in[57];
    const float* fc2W = (const float*)d_in[58]; const float* fc2b = (const float*)d_in[59];
    const float* fc3W = (const float*)d_in[60]; const float* fc3b = (const float*)d_in[61];

    // Workspace bump allocator
    char* ws = (char*)d_ws; size_t off = 0;
    auto alloc = [&](size_t bytes) -> void* {
        void* p = ws + off;
        off = (off + bytes + 255) & ~(size_t)255;
        return p;
    };
    float* featB[2]; for (int p = 0; p < 2; p++) featB[p] = (float*)alloc((size_t)B_ * 64 * N_ * 4);
    float*    latf = (float*)alloc((size_t)B_ * L0_ * CIN_ * 4);
    uint16_t* latb = (uint16_t*)alloc((size_t)B_ * L0_ * CIN_ * 2);
    float *oa[2], *ob[2], *oc[2];
    for (int p = 0; p < 2; p++) oa[p] = (float*)alloc((size_t)B_ * 64 * L0_ * 4);
    for (int p = 0; p < 2; p++) ob[p] = (float*)alloc((size_t)B_ * 64 * L1_ * 4);
    for (int p = 0; p < 2; p++) oc[p] = (float*)alloc((size_t)B_ * 64 * L2_ * 4);
    uint16_t* oc_sm[2]; for (int p = 0; p < 2; p++) oc_sm[p] = (uint16_t*)alloc((size_t)B_ * L2_ * 64 * 2);
    float*    cbuf = (float*)alloc((size_t)B_ * 32 * L2_ * 4);
    uint16_t* c_sm = (uint16_t*)alloc((size_t)B_ * L2_ * 32 * 2);
    float*    vbuf = (float*)alloc((size_t)B_ * 64 * L2_ * 4);
    float* pooled = (float*)alloc((size_t)B_ * 192 * 4);
    float* h1 = (float*)alloc((size_t)B_ * 2048 * 4);
    float* h2 = (float*)alloc((size_t)B_ * 2048 * 4);
    uint16_t* A_bcn[3]; uint16_t* W1b[3];
    for (int i = 0; i < 3; i++) {
        A_bcn[i] = (uint16_t*)alloc((size_t)64 * 1024 * 2);
        W1b[i]   = (uint16_t*)alloc((size_t)64 * 64 * 2);
    }
    uint16_t* A_c = (uint16_t*)alloc((size_t)32 * 1024 * 2);
    uint16_t* W2c = (uint16_t*)alloc((size_t)32 * 32 * 2);
    uint16_t* A_v = (uint16_t*)alloc((size_t)64 * 512 * 2);
    uint16_t* W2v = (uint16_t*)alloc((size_t)64 * 64 * 2);

    // Weight prep (fp32 -> bf16, blur weights flattened k = f*CIN + c)
    for (int i = 0; i < 3; i++) {
        prep_A<<<(64 * 1024 + 255) / 256, 256, 0, stream>>>(bcnWb[i], A_bcn[i], 64, CIN_, 1024);
        f32_to_bf16<<<(64 * 64 + 255) / 256, 256, 0, stream>>>(bcnW1[i], W1b[i], 64 * 64);
    }
    prep_A<<<(32 * 1024 + 255) / 256, 256, 0, stream>>>(Wc1, A_c, 32, 64, 1024);
    f32_to_bf16<<<(32 * 32 + 255) / 256, 256, 0, stream>>>(Wc2, W2c, 32 * 32);
    prep_A<<<(64 * 512 + 255) / 256, 256, 0, stream>>>(Wv1, A_v, 64, 32, 512);
    f32_to_bf16<<<(64 * 64 + 255) / 256, 256, 0, stream>>>(Wv2, W2v, 64 * 64);

    // Pointwise conv blocks
    for (int p = 0; p < 2; p++)
        feat_block_kernel<<<(B_ * N_) / 256, 256, 0, stream>>>(
            pc[p], cbW1, cbb1, cbW2, cbb2, cbW3, cbb3, featB[p]);

    // BCN stage runner: zero -> splat -> bf16 convert -> fused gather/WMMA blur
    size_t smBCN = smem_bytes(64, 1024);
    hipFuncSetAttribute((const void*)blur_kernel<CIN_, 64, 1024>,
                        hipFuncAttributeMaxDynamicSharedMemorySize, (int)smBCN);
    auto run_bcn = [&](int s, int p, const float* prevFeat, int M, int L,
                       const uint16_t* Ab, const float* bb1, const uint16_t* W2b,
                       const float* bb2, float* outp) {
        long latN = (long)B_ * L * CIN_;
        zero_f32<<<(int)((latN + 255) / 256), 256, 0, stream>>>(latf, latN);
        splat_kernel<<<(B_ * M + 255) / 256, 256, 0, stream>>>(
            el[s][p], prevFeat, bary[s][p], offs[s][p], latf, M, L);
        f32_to_bf16<<<(int)((latN + 255) / 256), 256, 0, stream>>>(latf, latb, latN);
        blur_kernel<CIN_, 64, 1024><<<B_ * (L / 64), 128, smBCN, stream>>>(
            latb, nbr[s][p], Ab, bb1, W2b, bb2, outp, nullptr, L);
    };

    for (int p = 0; p < 2; p++)
        run_bcn(0, p, featB[p], N_, L0_, A_bcn[0], bcnbb[0], W1b[0], bcnb1[0], oa[p]);
    for (int p = 0; p < 2; p++)
        run_bcn(1, p, oa[p], L0_, L1_, A_bcn[1], bcnbb[1], W1b[1], bcnb1[1], ob[p]);
    for (int p = 0; p < 2; p++)
        run_bcn(2, p, ob[p], L1_, L2_, A_bcn[2], bcnbb[2], W1b[2], bcnb1[2], oc[p]);

    // Correlation: prod = o1c * gather(o2c) folded into the gather, then blur.
    for (int p = 0; p < 2; p++)
        to_site_major_bf16<<<(B_ * 64 * L2_ + 255) / 256, 256, 0, stream>>>(
            oc[p], oc_sm[p], 64, L2_);

    size_t smC = smem_bytes(32, 1024);
    hipFuncSetAttribute((const void*)blur_kernel<64, 32, 1024>,
                        hipFuncAttributeMaxDynamicSharedMemorySize, (int)smC);
    blur_kernel<64, 32, 1024><<<B_ * (L2_ / 64), 128, smC, stream>>>(
        oc_sm[1], corr_idx2, A_c, bc1, W2c, bc2, cbuf, oc_sm[0], L2_);

    to_site_major_bf16<<<(B_ * 32 * L2_ + 255) / 256, 256, 0, stream>>>(cbuf, c_sm, 32, L2_);

    size_t smV = smem_bytes(64, 512);
    hipFuncSetAttribute((const void*)blur_kernel<32, 64, 512>,
                        hipFuncAttributeMaxDynamicSharedMemorySize, (int)smV);
    blur_kernel<32, 64, 512><<<B_ * (L2_ / 64), 128, smV, stream>>>(
        c_sm, corr_idx1, A_v, bv1, W2v, bv2, vbuf, nullptr, L2_);

    // Pool + FC head
    pool_kernel<<<B_ * 192, 256, 0, stream>>>(oc[0], oc[1], vbuf, pooled);
    fc_kernel<<<(B_ * 2048 + 255) / 256, 256, 0, stream>>>(pooled, fc1W, fc1b, h1, 192, 2048);
    fc_kernel<<<(B_ * 2048 + 255) / 256, 256, 0, stream>>>(h1, fc2W, fc2b, h2, 2048, 2048);
    fc3_kernel<<<1, 32, 0, stream>>>(h2, fc3W, fc3b, (float*)d_out);
}
